// MLA_87892210746097
// MI455X (gfx1250) — compile-verified, matrix operand order checked by reading, not run on hardware
//
#include <hip/hip_runtime.h>
#include <hip/hip_bf16.h>

// ---------------------------------------------------------------------------
// MHA forward (double-softmax variant) for gfx1250.
// All GEMMs run on v_wmma_f32_16x16x32_bf16 with fp32 emulated as bf16 hi/lo
// (3 WMMAs per 16x16x32 step).  fp32->bf16 splitting is hoisted OUT of the
// GEMM hot loops: operands are pre-split into bf16 "hi"/"lo" planes laid out
// so every fragment load is a contiguous 16B b128 load (B-side matrices are
// stored transposed).  Only the 512MB attention matrix is split in-loop.
// ---------------------------------------------------------------------------

constexpr int BATCH = 2;
constexpr int SEQ   = 2048;
constexpr int EMB   = 2048;
constexpr int NH    = 16;
constexpr int HD    = 128;                       // head dim
constexpr size_t PLN = (size_t)BATCH * NH * SEQ * HD;   // 8,388,608 elements
constexpr size_t WE  = (size_t)EMB * EMB;               // 4,194,304 elements

typedef __attribute__((ext_vector_type(16))) __bf16 v16bf;
typedef __attribute__((ext_vector_type(8)))  __bf16 v8bf;
typedef __attribute__((ext_vector_type(8)))  float  v8f;
typedef __attribute__((ext_vector_type(4)))  float  v4f;

// ---- fp32 -> bf16 hi/lo split (native RNE cvt / fma_mix instructions) -----
__device__ __forceinline__ void split_bf16(float x, __bf16& h, __bf16& l) {
    __bf16 hh = (__bf16)x;
    h = hh;
    l = (__bf16)(x - (float)hh);
}

// ---------------------------------------------------------------------------
// Fragment loaders.
// A 16x32 16-bit fragment: lanes 0-15 hold rows, K = {0..7,16..23};
// lanes 16-31 same rows, K + 8.  Per lane: two contiguous 8-element runs.
// ---------------------------------------------------------------------------

// bf16 plane, row-major: two b128 loads, no conversion.
__device__ __forceinline__ v16bf load_frag_bf16(const __bf16* __restrict__ p, int ld,
                                                int row0, int k0) {
    int lane = threadIdx.x & 31;
    const __bf16* base = p + (size_t)(row0 + (lane & 15)) * ld + k0 + ((lane >> 4) << 3);
    v8bf a = *(const v8bf*)(base);
    v8bf b = *(const v8bf*)(base + 16);
    __builtin_prefetch(base + 32, 0, 3);       // next k-step (global_prefetch_b8)
    return __builtin_shufflevector(a, b, 0, 1, 2, 3, 4, 5, 6, 7,
                                   8, 9, 10, 11, 12, 13, 14, 15);
}

// fp32 matrix, row-major: four b128 loads + in-register hi/lo split
// (used only for the attention matrix).
__device__ __forceinline__ void load_frag_f32split(const float* __restrict__ p, int ld,
                                                   int row0, int k0,
                                                   v16bf& hi, v16bf& lo) {
    int lane = threadIdx.x & 31;
    const float* base = p + (size_t)(row0 + (lane & 15)) * ld + k0 + ((lane >> 4) << 3);
    v4f c0 = *(const v4f*)(base + 0);
    v4f c1 = *(const v4f*)(base + 4);
    v4f c2 = *(const v4f*)(base + 16);
    v4f c3 = *(const v4f*)(base + 20);
    __builtin_prefetch(base + 32, 0, 3);
    float xs[16];
#pragma unroll
    for (int e = 0; e < 4; ++e) { xs[e] = c0[e]; xs[4 + e] = c1[e]; xs[8 + e] = c2[e]; xs[12 + e] = c3[e]; }
#pragma unroll
    for (int e = 0; e < 16; ++e) {
        __bf16 h, l;
        split_bf16(xs[e], h, l);
        hi[e] = h;
        lo[e] = l;
    }
}

// 3-product bf16 emulation of an fp32 16x16x32 MAC step.
__device__ __forceinline__ v8f wmma3(v8f acc, v16bf ah, v16bf al, v16bf bh, v16bf bl) {
    acc = __builtin_amdgcn_wmma_f32_16x16x32_bf16(false, al, false, bh, (short)0, acc, false, false);
    acc = __builtin_amdgcn_wmma_f32_16x16x32_bf16(false, ah, false, bl, (short)0, acc, false, false);
    acc = __builtin_amdgcn_wmma_f32_16x16x32_bf16(false, ah, false, bh, (short)0, acc, false, false);
    return acc;
}

// ---------------------------------------------------------------------------
// Pre-pass conversion kernels.
// ---------------------------------------------------------------------------

// Elementwise split: f32 plane -> bf16 hi/lo planes (same layout).
__global__ void split_plane_kernel(const float* __restrict__ src,
                                   __bf16* __restrict__ oh, __bf16* __restrict__ ol) {
    size_t i = (size_t)blockIdx.x * 256 + threadIdx.x;
    __bf16 h, l;
    split_bf16(src[i], h, l);
    oh[i] = h;
    ol[i] = l;
}

// Transpose + split: W[K][N] (row-major, 2048x2048) -> WT[n][k] hi/lo planes.
__global__ void splitT_kernel(const float* __restrict__ W,
                              __bf16* __restrict__ oh, __bf16* __restrict__ ol) {
    size_t i = (size_t)blockIdx.x * 256 + threadIdx.x;   // over 2048*2048
    int n = (int)(i >> 11);
    int k = (int)(i & 2047);
    __bf16 h, l;
    split_bf16(W[(size_t)k * EMB + n], h, l);
    oh[i] = h;
    ol[i] = l;
}

// ---------------------------------------------------------------------------
// proj_kernel: C = A(4096 x 2048) @ B + bias, A/B given as bf16 hi/lo planes,
// B pre-transposed (BT[n][k]).  One wave = one 32x32 macro-tile, 4 waves/blk.
//   mode 0: fout[m][n] = C (fp32, final output projection)
//   mode 1: split C to vT planes   [B][H][D][S]      (V projection)
//   mode 2: RoPE, split C to q/k planes [B][H][S][D] (Q/K projections)
// ---------------------------------------------------------------------------
__global__ void proj_kernel(const __bf16* __restrict__ Ah, const __bf16* __restrict__ Al,
                            const __bf16* __restrict__ BTh, const __bf16* __restrict__ BTl,
                            const float* __restrict__ bias,
                            const float* __restrict__ sinp, const float* __restrict__ cosp,
                            float* __restrict__ fout,
                            __bf16* __restrict__ oh, __bf16* __restrict__ ol,
                            int mode) {
    const int N = EMB, K = EMB;
    int strip = blockIdx.x * 4 + (threadIdx.x >> 5);
    int ntn   = N / 32;                     // 64
    int tm = strip / ntn, tn = strip % ntn;
    int rowB = tm * 32, colB = tn * 32;

    v8f acc[2][2] = {};
    for (int k0 = 0; k0 < K; k0 += 32) {
        v16bf ah0 = load_frag_bf16(Ah, K, rowB,      k0);
        v16bf ah1 = load_frag_bf16(Ah, K, rowB + 16, k0);
        v16bf al0 = load_frag_bf16(Al, K, rowB,      k0);
        v16bf al1 = load_frag_bf16(Al, K, rowB + 16, k0);
        v16bf bh0 = load_frag_bf16(BTh, K, colB,      k0);
        v16bf bh1 = load_frag_bf16(BTh, K, colB + 16, k0);
        v16bf bl0 = load_frag_bf16(BTl, K, colB,      k0);
        v16bf bl1 = load_frag_bf16(BTl, K, colB + 16, k0);
        acc[0][0] = wmma3(acc[0][0], ah0, al0, bh0, bl0);
        acc[0][1] = wmma3(acc[0][1], ah0, al0, bh1, bl1);
        acc[1][0] = wmma3(acc[1][0], ah1, al1, bh0, bl0);
        acc[1][1] = wmma3(acc[1][1], ah1, al1, bh1, bl1);
    }

    int lane = threadIdx.x & 31;
#pragma unroll
    for (int i = 0; i < 2; ++i) {
#pragma unroll
        for (int j = 0; j < 2; ++j) {
            int n     = colB + j * 16 + (lane & 15);
            int rbase = rowB + i * 16 + ((lane >> 4) << 3);
            float bv  = bias[n];
#pragma unroll
            for (int r = 0; r < 8; ++r) {
                int   m   = rbase + r;
                float val = acc[i][j][r] + bv;
                if (mode == 0) {
                    fout[(size_t)m * N + n] = val;
                } else {
                    int b = m >> 11, s = m & (SEQ - 1);
                    int h = n >> 7,  d = n & (HD - 1);
                    if (mode == 2) {
                        // RoPE: pairs (d, d^1) sit in adjacent lanes of C.
                        float partner = __shfl_xor(val, 1, 32);
                        float rot     = (d & 1) ? partner : -partner;
                        size_t sidx   = ((size_t)h * SEQ + s) * HD + d;
                        val = val * cosp[sidx] + rot * sinp[sidx];
                    }
                    __bf16 vh, vl;
                    split_bf16(val, vh, vl);
                    size_t oidx;
                    if (mode == 1)   // transposed: [B][H][D][S]
                        oidx = (((size_t)b * NH + h) * HD + d) * SEQ + s;
                    else             // [B][H][S][D]
                        oidx = (((size_t)b * NH + h) * SEQ + s) * HD + d;
                    oh[oidx] = vh;
                    ol[oidx] = vl;
                }
            }
        }
    }
}

// ---------------------------------------------------------------------------
// qk_kernel: per (b,h) raw logits  C[S x S] = Q[S x D] @ K^T   (fp32 out)
// Q,K given as bf16 hi/lo planes [B][H][S][D]; B-fragment of K^T is the
// A-style fragment of K (free transpose).  Zero conversions in the loop.
// ---------------------------------------------------------------------------
__global__ void qk_kernel(const __bf16* __restrict__ Qh, const __bf16* __restrict__ Ql,
                          const __bf16* __restrict__ Kh, const __bf16* __restrict__ Kl,
                          float* __restrict__ attn) {
    int bh = blockIdx.y;
    size_t off = (size_t)bh * SEQ * HD;
    const __bf16* qh = Qh + off;
    const __bf16* ql = Ql + off;
    const __bf16* kh = Kh + off;
    const __bf16* kl = Kl + off;
    float* outp = attn + (size_t)bh * SEQ * SEQ;

    int strip = blockIdx.x * 4 + (threadIdx.x >> 5);
    int ntn   = SEQ / 32;                   // 64
    int tm = strip / ntn, tn = strip % ntn;
    int rowB = tm * 32, colB = tn * 32;

    v8f acc[2][2] = {};
#pragma unroll
    for (int k0 = 0; k0 < HD; k0 += 32) {
        v16bf ah0 = load_frag_bf16(qh, HD, rowB,      k0);
        v16bf ah1 = load_frag_bf16(qh, HD, rowB + 16, k0);
        v16bf al0 = load_frag_bf16(ql, HD, rowB,      k0);
        v16bf al1 = load_frag_bf16(ql, HD, rowB + 16, k0);
        v16bf bh0 = load_frag_bf16(kh, HD, colB,      k0);
        v16bf bh1 = load_frag_bf16(kh, HD, colB + 16, k0);
        v16bf bl0 = load_frag_bf16(kl, HD, colB,      k0);
        v16bf bl1 = load_frag_bf16(kl, HD, colB + 16, k0);
        acc[0][0] = wmma3(acc[0][0], ah0, al0, bh0, bl0);
        acc[0][1] = wmma3(acc[0][1], ah0, al0, bh1, bl1);
        acc[1][0] = wmma3(acc[1][0], ah1, al1, bh0, bl0);
        acc[1][1] = wmma3(acc[1][1], ah1, al1, bh1, bl1);
    }

    int lane = threadIdx.x & 31;
#pragma unroll
    for (int i = 0; i < 2; ++i) {
#pragma unroll
        for (int j = 0; j < 2; ++j) {
            int n     = colB + j * 16 + (lane & 15);
            int rbase = rowB + i * 16 + ((lane >> 4) << 3);
#pragma unroll
            for (int r = 0; r < 8; ++r)
                outp[(size_t)(rbase + r) * SEQ + n] = acc[i][j][r];
        }
    }
}

// ---------------------------------------------------------------------------
// row_softmax_kernel: in-place softmax over rows of length SEQ.
//   masked=1: val = logit*scale, then mask==0 -> -9e15 (first softmax)
//   masked=0: plain softmax (second softmax)
// ---------------------------------------------------------------------------
__device__ __forceinline__ float warp_max(float v) {
#pragma unroll
    for (int o = 16; o > 0; o >>= 1) v = fmaxf(v, __shfl_xor(v, o, 32));
    return v;
}
__device__ __forceinline__ float warp_sum(float v) {
#pragma unroll
    for (int o = 16; o > 0; o >>= 1) v += __shfl_xor(v, o, 32);
    return v;
}

__global__ void row_softmax_kernel(float* __restrict__ attn, const int* __restrict__ mask,
                                   float scale, int masked) {
    constexpr int TPB = 256;
    constexpr int PER = SEQ / TPB;          // 8
    __shared__ float sh[TPB / 32];

    int row   = blockIdx.x;                 // 0 .. B*H*S-1
    float* rp = attn + (size_t)row * SEQ;
    const int* mp = nullptr;
    if (masked) {
        int sq = row & (SEQ - 1);
        int b  = row / (NH * SEQ);
        mp = mask + ((size_t)b * SEQ + sq) * SEQ;
    }

    float v[PER];
#pragma unroll
    for (int i = 0; i < PER; ++i) {
        int sk  = threadIdx.x + i * TPB;
        float x = rp[sk] * scale;
        if (masked && mp[sk] == 0) x = -9.0e15f;
        v[i] = x;
    }

    float m = -3.402823466e38f;
#pragma unroll
    for (int i = 0; i < PER; ++i) m = fmaxf(m, v[i]);
    m = warp_max(m);
    if ((threadIdx.x & 31) == 0) sh[threadIdx.x >> 5] = m;
    __syncthreads();
    float bm = sh[0];
#pragma unroll
    for (int w = 1; w < TPB / 32; ++w) bm = fmaxf(bm, sh[w]);
    __syncthreads();

    float sum = 0.f;
#pragma unroll
    for (int i = 0; i < PER; ++i) { v[i] = __expf(v[i] - bm); sum += v[i]; }
    sum = warp_sum(sum);
    if ((threadIdx.x & 31) == 0) sh[threadIdx.x >> 5] = sum;
    __syncthreads();
    float bs = 0.f;
#pragma unroll
    for (int w = 0; w < TPB / 32; ++w) bs += sh[w];

    float inv = 1.0f / bs;
#pragma unroll
    for (int i = 0; i < PER; ++i) rp[threadIdx.x + i * TPB] = v[i] * inv;
}

// ---------------------------------------------------------------------------
// av_kernel: per (b,h)  C[S x D] = A[S x S] @ V
//   A: fp32 attention matrix (split in-loop).
//   V: bf16 hi/lo planes, pre-transposed [B][H][D][S] (contiguous B-frags).
//   out_mode 0: split C to valsT planes [B][H][D][S]
//   out_mode 1: split C to ao planes    [B][S][E]
// ---------------------------------------------------------------------------
__global__ void av_kernel(const float* __restrict__ Am,
                          const __bf16* __restrict__ VTh, const __bf16* __restrict__ VTl,
                          __bf16* __restrict__ oh, __bf16* __restrict__ ol,
                          int out_mode) {
    int bh = blockIdx.y;
    int b = bh >> 4, h = bh & 15;
    const float*  a   = Am  + (size_t)bh * SEQ * SEQ;
    const __bf16* vth = VTh + (size_t)bh * HD * SEQ;
    const __bf16* vtl = VTl + (size_t)bh * HD * SEQ;

    int strip = blockIdx.x * 4 + (threadIdx.x >> 5);
    int ntn   = HD / 32;                    // 4
    int tm = strip / ntn, tn = strip % ntn;
    int rowB = tm * 32, colB = tn * 32;

    v8f acc[2][2] = {};
    for (int k0 = 0; k0 < SEQ; k0 += 32) {
        v16bf ah0, al0, ah1, al1;
        load_frag_f32split(a, SEQ, rowB,      k0, ah0, al0);
        load_frag_f32split(a, SEQ, rowB + 16, k0, ah1, al1);
        v16bf bh0 = load_frag_bf16(vth, SEQ, colB,      k0);
        v16bf bh1 = load_frag_bf16(vth, SEQ, colB + 16, k0);
        v16bf bl0 = load_frag_bf16(vtl, SEQ, colB,      k0);
        v16bf bl1 = load_frag_bf16(vtl, SEQ, colB + 16, k0);
        acc[0][0] = wmma3(acc[0][0], ah0, al0, bh0, bl0);
        acc[0][1] = wmma3(acc[0][1], ah0, al0, bh1, bl1);
        acc[1][0] = wmma3(acc[1][0], ah1, al1, bh0, bl0);
        acc[1][1] = wmma3(acc[1][1], ah1, al1, bh1, bl1);
    }

    int lane = threadIdx.x & 31;
#pragma unroll
    for (int i = 0; i < 2; ++i) {
#pragma unroll
        for (int j = 0; j < 2; ++j) {
            int d     = colB + j * 16 + (lane & 15);
            int rbase = rowB + i * 16 + ((lane >> 4) << 3);
#pragma unroll
            for (int r = 0; r < 8; ++r) {
                int sq = rbase + r;
                __bf16 vh, vl;
                split_bf16(acc[i][j][r], vh, vl);
                size_t oidx;
                if (out_mode == 0)   // valsT: [B][H][D][S]
                    oidx = ((size_t)bh * HD + d) * SEQ + sq;
                else                 // ao: [B][S][E]
                    oidx = ((size_t)b * SEQ + sq) * EMB + (size_t)h * HD + d;
                oh[oidx] = vh;
                ol[oidx] = vl;
            }
        }
    }
}

// ---------------------------------------------------------------------------
extern "C" void kernel_launch(void* const* d_in, const int* in_sizes, int n_in,
                              void* d_out, int out_size, void* d_ws, size_t ws_size,
                              hipStream_t stream) {
    (void)in_sizes; (void)n_in; (void)out_size; (void)ws_size;

    const float* x    = (const float*)d_in[0];
    const float* sinp = (const float*)d_in[1];
    const float* cosp = (const float*)d_in[2];
    const int*   mask = (const int*)d_in[3];
    const float* Wq = (const float*)d_in[4];  const float* bq = (const float*)d_in[5];
    const float* Wk = (const float*)d_in[6];  const float* bk = (const float*)d_in[7];
    const float* Wv = (const float*)d_in[8];  const float* bv = (const float*)d_in[9];
    const float* Wo = (const float*)d_in[10]; const float* bo = (const float*)d_in[11];

    float* out   = (float*)d_out;                          // [B][S][E]
    float* attnW = out + (size_t)BATCH * SEQ * EMB;        // [B][H][S][S] (2nd output)

    // bf16 hi/lo plane workspace (256 MiB total)
    __bf16* p = (__bf16*)d_ws;
    __bf16 *xh   = p;            __bf16 *xl   = xh + PLN;     p += 2 * PLN;
    __bf16 *WqTh = p;            __bf16 *WqTl = WqTh + WE;    p += 2 * WE;
    __bf16 *WkTh = p;            __bf16 *WkTl = WkTh + WE;    p += 2 * WE;
    __bf16 *WvTh = p;            __bf16 *WvTl = WvTh + WE;    p += 2 * WE;
    __bf16 *WoTh = p;            __bf16 *WoTl = WoTh + WE;    p += 2 * WE;
    __bf16 *qhp  = p;            __bf16 *qlp  = qhp + PLN;    p += 2 * PLN;
    __bf16 *khp  = p;            __bf16 *klp  = khp + PLN;    p += 2 * PLN;
    __bf16 *vTh  = p;            __bf16 *vTl  = vTh + PLN;    p += 2 * PLN;
    __bf16 *valsTh = p;          __bf16 *valsTl = valsTh + PLN; p += 2 * PLN;
    __bf16 *aoh  = p;            __bf16 *aol  = aoh + PLN;    p += 2 * PLN;

    const float scale = 0.088388347648318447f;             // 1/sqrt(128)

    // --- pre-pass: split x; transpose+split the four weight matrices -------
    split_plane_kernel<<<(unsigned)(PLN / 256), 256, 0, stream>>>(x, xh, xl);
    splitT_kernel<<<(unsigned)(WE / 256), 256, 0, stream>>>(Wq, WqTh, WqTl);
    splitT_kernel<<<(unsigned)(WE / 256), 256, 0, stream>>>(Wk, WkTh, WkTl);
    splitT_kernel<<<(unsigned)(WE / 256), 256, 0, stream>>>(Wv, WvTh, WvTl);
    splitT_kernel<<<(unsigned)(WE / 256), 256, 0, stream>>>(Wo, WoTh, WoTl);

    // --- Q/K/V projections (+RoPE on Q,K) ----------------------------------
    proj_kernel<<<2048, 128, 0, stream>>>(xh, xl, WqTh, WqTl, bq, sinp, cosp,
                                          nullptr, qhp, qlp, 2);
    proj_kernel<<<2048, 128, 0, stream>>>(xh, xl, WkTh, WkTl, bk, sinp, cosp,
                                          nullptr, khp, klp, 2);
    proj_kernel<<<2048, 128, 0, stream>>>(xh, xl, WvTh, WvTl, bv, nullptr, nullptr,
                                          nullptr, vTh, vTl, 1);

    // --- logits = Q K^T (raw) into the attention_weights output region -----
    qk_kernel<<<dim3(1024, 32), 128, 0, stream>>>(qhp, qlp, khp, klp, attnW);

    // --- attention = softmax(mask ? logits/sqrt(D) : -9e15)   (in place) ---
    row_softmax_kernel<<<BATCH * NH * SEQ, 256, 0, stream>>>(attnW, mask, scale, 1);

    // --- values = attention @ V  -> valsT planes ---------------------------
    av_kernel<<<dim3(64, 32), 128, 0, stream>>>(attnW, vTh, vTl, valsTh, valsTl, 0);

    // --- attention_weights = softmax(attention)  (in place, final output) --
    row_softmax_kernel<<<BATCH * NH * SEQ, 256, 0, stream>>>(attnW, nullptr, 1.0f, 0);

    // --- attn_output = attention_weights @ values -> ao planes [B][S][E] ---
    av_kernel<<<dim3(64, 32), 128, 0, stream>>>(attnW, valsTh, valsTl, aoh, aol, 1);

    // --- out = attn_output @ Wo + bo ---------------------------------------
    proj_kernel<<<2048, 128, 0, stream>>>(aoh, aol, WoTh, WoTl, bo, nullptr, nullptr,
                                          out, nullptr, nullptr, 0);
}
